// RelativePositionSelfAttention_30906584662592
// MI455X (gfx1250) — matrix operand order repaired and checked by reading, not compiled, not run on hardware
//
#include <hip/hip_runtime.h>
#include <hip/hip_bf16.h>

#define D_MODEL   1024
#define NUM_HEADS 16
#define MAX_LEN   500
#define D_K       64
#define B_SZ      8
#define L_SEQ     500
#define L_PAD     512

typedef unsigned short u16;
typedef __attribute__((ext_vector_type(8)))  u16    u16x8;
typedef __attribute__((ext_vector_type(16))) u16    u16x16;
typedef __attribute__((ext_vector_type(16))) __bf16 v16bf;
typedef __attribute__((ext_vector_type(8)))  float  v8f;
typedef __attribute__((ext_vector_type(4)))  unsigned int uint32x4;
typedef __attribute__((ext_vector_type(8)))  int    int32x8;
typedef __attribute__((ext_vector_type(4)))  int    int32x4;

// fp32 -> bf16 with round-to-nearest-even
__device__ __forceinline__ u16 f2bf(float f) {
  unsigned u = __builtin_bit_cast(unsigned, f);
  u += 0x7fffu + ((u >> 16) & 1u);
  return (u16)(u >> 16);
}

__device__ __forceinline__ v16bf mk_frag(u16x8 lo, u16x8 hi) {
  u16x16 w = __builtin_shufflevector(lo, hi, 0,1,2,3,4,5,6,7,8,9,10,11,12,13,14,15);
  return __builtin_bit_cast(v16bf, w);
}

__device__ __forceinline__ v8f wmma_bf16(v16bf a, v16bf b, v8f c) {
  return __builtin_amdgcn_wmma_f32_16x16x32_bf16(false, a, false, b, (short)0, c,
                                                 false, false);
}

// Tensor DMA descriptor builders (CDNA5 ISA).
// Group 0: [1:0]=count=1, [63:32]=lds_addr, [120:64]=global_addr, [127:126]=type(2)
__device__ __forceinline__ uint32x4 tdm_group0(unsigned lds_off,
                                               unsigned long long gaddr) {
  uint32x4 g;
  g[0] = 1u;                                      // count=1, user mode
  g[1] = lds_off;                                 // LDS byte address
  g[2] = (unsigned)(gaddr & 0xffffffffu);         // global_addr[31:0]
  g[3] = (unsigned)((gaddr >> 32) & 0x01ffffffu)  // global_addr[56:32]
       | 0x80000000u;                             // type = 2 ("image")
  return g;
}

// Group 1: data_size=1(2B); tensor_dim0/1, tile_dim0/1, tensor_dim0_stride
__device__ __forceinline__ int32x8 tdm_group1(unsigned td0, unsigned td1,
                                              unsigned tile0, unsigned tile1,
                                              unsigned long long stride0) {
  int32x8 g;
  g[0] = (int)(1u << 16);                              // data_size = 2 bytes
  g[1] = (int)((td0 & 0xffffu) << 16);                 // tensor_dim0[15:0]
  g[2] = (int)(((td0 >> 16) & 0xffffu) | ((td1 & 0xffffu) << 16));
  g[3] = (int)(((td1 >> 16) & 0xffffu) | (tile0 << 16)); // tile_dim0
  g[4] = (int)(tile1);                                 // tile_dim1 (tile_dim2=0)
  g[5] = (int)(stride0 & 0xffffffffu);                 // tensor_dim0_stride lo
  g[6] = (int)((stride0 >> 32) & 0xffffu);             // stride0 hi, stride1=0
  g[7] = 0;
  return g;
}

// ---------------------------------------------------------------------------
// Kernel 1: fused QKV projection.  Y = X @ W + bias, output in bf16.
//   z=0: Q -> Qb[(b*16+h)][lrow(512)][dk(64)]   row-major, padded
//   z=1: K -> Kb same layout
//   z=2: V -> Vt[(b*16+h)][dk(64)][lrow(512)]   TRANSPOSED (for B-fragments)
// ---------------------------------------------------------------------------
__global__ __launch_bounds__(128) void proj_kernel(
    const float* __restrict__ q_in, const float* __restrict__ k_in,
    const float* __restrict__ v_in,
    const float* __restrict__ Wq, const float* __restrict__ Wk,
    const float* __restrict__ Wv,
    const float* __restrict__ bq, const float* __restrict__ bk,
    const float* __restrict__ bv,
    u16* __restrict__ Qb, u16* __restrict__ Kb, u16* __restrict__ Vt)
{
  const int z = blockIdx.z;
  const float* X    = (z == 0) ? q_in : (z == 1) ? k_in : v_in;
  const float* W    = (z == 0) ? Wq   : (z == 1) ? Wk   : Wv;
  const float* bias = (z == 0) ? bq   : (z == 1) ? bk   : bv;

  const int lane = threadIdx.x & 31;
  const int wave = threadIdx.x >> 5;
  const int hf   = lane >> 4;
  const int ln   = lane & 15;

  const int m0   = blockIdx.x * 16;
  const int col0 = (blockIdx.y * 4 + wave) * 16;
  const int n    = col0 + ln;
  const int arow = m0 + ln;

  v8f acc = {};

  for (int k0 = 0; k0 < D_MODEL; k0 += 32) {
    const float* xp = X + (size_t)arow * D_MODEL + k0;
    float4 a0 = *(const float4*)(xp + 8 * hf);
    float4 a1 = *(const float4*)(xp + 8 * hf + 4);
    float4 a2 = *(const float4*)(xp + 16 + 8 * hf);
    float4 a3 = *(const float4*)(xp + 16 + 8 * hf + 4);
    u16x16 aw;
    aw[0]  = f2bf(a0.x); aw[1]  = f2bf(a0.y); aw[2]  = f2bf(a0.z); aw[3]  = f2bf(a0.w);
    aw[4]  = f2bf(a1.x); aw[5]  = f2bf(a1.y); aw[6]  = f2bf(a1.z); aw[7]  = f2bf(a1.w);
    aw[8]  = f2bf(a2.x); aw[9]  = f2bf(a2.y); aw[10] = f2bf(a2.z); aw[11] = f2bf(a2.w);
    aw[12] = f2bf(a3.x); aw[13] = f2bf(a3.y); aw[14] = f2bf(a3.z); aw[15] = f2bf(a3.w);
    v16bf afrag = __builtin_bit_cast(v16bf, aw);

    const float* wp = W + (size_t)(k0 + 8 * hf) * D_MODEL + n;
    u16x16 bw;
#pragma unroll
    for (int e = 0; e < 16; ++e) bw[e] = f2bf(wp[(size_t)(e + (e >= 8 ? 8 : 0)) * D_MODEL]);
    v16bf bfrag = __builtin_bit_cast(v16bf, bw);

    acc = wmma_bf16(afrag, bfrag, acc);
  }

  const float bn  = bias[n];
  const int   h   = col0 >> 6;
  const int   cih = (col0 & 63) + ln;

#pragma unroll
  for (int r = 0; r < 8; ++r) {
    const int mg   = m0 + r + 8 * hf;
    const int bidx = mg / L_SEQ;
    const int lrow = mg - bidx * L_SEQ;
    const int bh   = bidx * NUM_HEADS + h;
    const u16 val  = f2bf(acc[r] + bn);
    if (z == 0) {
      Qb[((size_t)bh * L_PAD + lrow) * D_K + cih] = val;
    } else if (z == 1) {
      Kb[((size_t)bh * L_PAD + lrow) * D_K + cih] = val;
    } else {
      Vt[((size_t)bh * D_K + cih) * L_PAD + lrow] = val;
    }
  }
}

// ---------------------------------------------------------------------------
// Kernel 2: fused flash attention with relative-position bias.
// 8 waves/block share one (b,h); K/V tiles staged into LDS once per block
// via the Tensor Data Mover, then all waves read WMMA fragments from LDS.
// ---------------------------------------------------------------------------
__global__ __launch_bounds__(256) void attn_kernel(
    const u16* __restrict__ Qb, const u16* __restrict__ Kb,
    const u16* __restrict__ Vt, const float* __restrict__ rel,
    float* __restrict__ out)
{
  __shared__ __attribute__((aligned(32))) u16 Ktile[32 * 64];   // [key][dk]
  __shared__ __attribute__((aligned(32))) u16 Vtile[64 * 32];   // [dk][key]
  __shared__ __attribute__((aligned(32))) u16 pLds[8][16 * 32];

  const int lane = threadIdx.x & 31;
  const int wave = threadIdx.x >> 5;
  const int hf   = lane >> 4;
  const int ln   = lane & 15;

  const int bh    = blockIdx.y;
  const int b     = bh >> 4;
  const int h     = bh & 15;
  const int qtile = blockIdx.x * 8 + wave;
  const int qbase = qtile * 16;

  const u16*   Qp   = Qb + (size_t)bh * L_PAD * D_K;
  const u16*   Kp   = Kb + (size_t)bh * L_PAD * D_K;
  const u16*   Vp   = Vt + (size_t)bh * D_K * L_PAD;
  const float* relh = rel + (size_t)h * (2 * MAX_LEN - 1);

  const unsigned kLdsOff = (unsigned)(size_t)(&Ktile[0]);
  const unsigned vLdsOff = (unsigned)(size_t)(&Vtile[0]);

  // Q A-fragments (16x64 split into two 16x32 depth windows), loaded once.
  const u16* qrow = Qp + (size_t)(qbase + ln) * D_K;
  v16bf aq0 = mk_frag(*(const u16x8*)(qrow + 16 * hf),
                      *(const u16x8*)(qrow + 16 * hf + 8));
  v16bf aq1 = mk_frag(*(const u16x8*)(qrow + 32 + 16 * hf),
                      *(const u16x8*)(qrow + 32 + 16 * hf + 8));

  v8f acc0 = {}, acc1 = {}, acc2 = {}, acc3 = {};
  float mrow[8], lsum[8];
#pragma unroll
  for (int r = 0; r < 8; ++r) { mrow[r] = -3.0e38f; lsum[r] = 0.0f; }

  const int32x4 gz4 = {0, 0, 0, 0};
  const int32x8 gz8 = {0, 0, 0, 0, 0, 0, 0, 0};

  for (int kb = 0; kb < L_PAD / 32; ++kb) {
    const int kbase = kb * 32;

    __syncthreads();   // all waves done reading previous tiles
    if (wave == 0) {
      // K tile: 32 rows x 64 elems, contiguous rows (stride 64 elems)
      uint32x4 kg0 = tdm_group0(kLdsOff,
                                (unsigned long long)(Kp + (size_t)kbase * D_K));
      int32x8  kg1 = tdm_group1(/*td0=*/64, /*td1=*/32, /*tile0=*/64,
                                /*tile1=*/32, /*stride0=*/64);
      __builtin_amdgcn_tensor_load_to_lds(kg0, kg1, gz4, gz4, gz8, 0);
      // V tile: 64 rows x 32 elems, row stride 512 elems (Vt is [dk][L_PAD])
      uint32x4 vg0 = tdm_group0(vLdsOff,
                                (unsigned long long)(Vp + kbase));
      int32x8  vg1 = tdm_group1(/*td0=*/32, /*td1=*/64, /*tile0=*/32,
                                /*tile1=*/64, /*stride0=*/512);
      __builtin_amdgcn_tensor_load_to_lds(vg0, vg1, gz4, gz4, gz8, 0);
      __builtin_amdgcn_s_wait_tensorcnt(0);
    }
    __syncthreads();   // tiles visible to all waves

    // ---- S = Q @ K^T : two 16x16 tiles from LDS K tile ----
    const u16* kr0 = &Ktile[(size_t)ln * D_K];             // keys kbase+ln
    const u16* kr1 = kr0 + 16 * D_K;                       // keys kbase+16+ln
    v16bf bk00 = mk_frag(*(const u16x8*)(kr0 + 16 * hf),
                         *(const u16x8*)(kr0 + 16 * hf + 8));
    v16bf bk01 = mk_frag(*(const u16x8*)(kr0 + 32 + 16 * hf),
                         *(const u16x8*)(kr0 + 32 + 16 * hf + 8));
    v16bf bk10 = mk_frag(*(const u16x8*)(kr1 + 16 * hf),
                         *(const u16x8*)(kr1 + 16 * hf + 8));
    v16bf bk11 = mk_frag(*(const u16x8*)(kr1 + 32 + 16 * hf),
                         *(const u16x8*)(kr1 + 32 + 16 * hf + 8));

    v8f s0 = {}, s1 = {};
    s0 = wmma_bf16(aq0, bk00, s0);
    s0 = wmma_bf16(aq1, bk01, s0);
    s1 = wmma_bf16(aq0, bk10, s1);
    s1 = wmma_bf16(aq1, bk11, s1);

    // ---- scale + relative-position bias + mask + online softmax ----
    float p0[8], p1[8];
#pragma unroll
    for (int r = 0; r < 8; ++r) {
      const int qi  = qbase + r + 8 * hf;
      const int k0i = kbase + ln;
      const int k1i = k0i + 16;
      float v0 = s0[r] * 0.125f;
      float v1 = s1[r] * 0.125f;
      int i0 = qi - k0i + (MAX_LEN - 1);
      int i1 = qi - k1i + (MAX_LEN - 1);
      i0 = i0 < 0 ? 0 : (i0 > 2 * MAX_LEN - 2 ? 2 * MAX_LEN - 2 : i0);
      i1 = i1 < 0 ? 0 : (i1 > 2 * MAX_LEN - 2 ? 2 * MAX_LEN - 2 : i1);
      v0 += relh[i0];
      v1 += relh[i1];
      if (k0i >= L_SEQ) v0 = -1.0e30f;
      if (k1i >= L_SEQ) v1 = -1.0e30f;

      float mx = fmaxf(v0, v1);
      mx = fmaxf(mx, __shfl_xor(mx, 1));
      mx = fmaxf(mx, __shfl_xor(mx, 2));
      mx = fmaxf(mx, __shfl_xor(mx, 4));
      mx = fmaxf(mx, __shfl_xor(mx, 8));
      const float mnew  = fmaxf(mrow[r], mx);
      const float alpha = __expf(mrow[r] - mnew);
      mrow[r] = mnew;

      const float e0 = __expf(v0 - mnew);
      const float e1 = __expf(v1 - mnew);
      float rs = e0 + e1;
      rs += __shfl_xor(rs, 1);
      rs += __shfl_xor(rs, 2);
      rs += __shfl_xor(rs, 4);
      rs += __shfl_xor(rs, 8);
      lsum[r] = lsum[r] * alpha + rs;
      acc0[r] *= alpha; acc1[r] *= alpha; acc2[r] *= alpha; acc3[r] *= alpha;
      p0[r] = e0; p1[r] = e1;
    }

    // ---- remap P from C-layout to A-layout through per-wave LDS slice ----
    u16* slice = pLds[wave];
    __builtin_amdgcn_wave_barrier();
#pragma unroll
    for (int r = 0; r < 8; ++r) {
      const int row = r + 8 * hf;
      slice[row * 32 + ln]      = f2bf(p0[r]);
      slice[row * 32 + 16 + ln] = f2bf(p1[r]);
    }
    __builtin_amdgcn_wave_barrier();   // DS ops are in-order per wave
    const u16* prow = slice + ln * 32;
    v16bf pf = mk_frag(*(const u16x8*)(prow + 16 * hf),
                       *(const u16x8*)(prow + 16 * hf + 8));
    __builtin_amdgcn_wave_barrier();

    // ---- O += P @ V : four 16x16 output tiles from LDS V tile ----
    const u16* vb = &Vtile[(size_t)ln * 32 + 16 * hf];
    v16bf bv0 = mk_frag(*(const u16x8*)(vb),
                        *(const u16x8*)(vb + 8));
    v16bf bv1 = mk_frag(*(const u16x8*)(vb + 16 * 32),
                        *(const u16x8*)(vb + 16 * 32 + 8));
    v16bf bv2 = mk_frag(*(const u16x8*)(vb + 32 * 32),
                        *(const u16x8*)(vb + 32 * 32 + 8));
    v16bf bv3 = mk_frag(*(const u16x8*)(vb + 48 * 32),
                        *(const u16x8*)(vb + 48 * 32 + 8));
    acc0 = wmma_bf16(pf, bv0, acc0);
    acc1 = wmma_bf16(pf, bv1, acc1);
    acc2 = wmma_bf16(pf, bv2, acc2);
    acc3 = wmma_bf16(pf, bv3, acc3);
  }

  // ---- normalize and store (fp32 output, (B, L, H*dk)) ----
#pragma unroll
  for (int r = 0; r < 8; ++r) {
    const int row = qbase + r + 8 * hf;
    if (row < L_SEQ) {
      const float inv = 1.0f / lsum[r];
      float* op = out + ((size_t)(b * L_SEQ + row) * D_MODEL) + h * D_K;
      op[ln]      = acc0[r] * inv;
      op[16 + ln] = acc1[r] * inv;
      op[32 + ln] = acc2[r] * inv;
      op[48 + ln] = acc3[r] * inv;
    }
  }
}

extern "C" void kernel_launch(void* const* d_in, const int* in_sizes, int n_in,
                              void* d_out, int out_size, void* d_ws, size_t ws_size,
                              hipStream_t stream) {
  const float* query = (const float*)d_in[0];
  const float* key   = (const float*)d_in[1];
  const float* value = (const float*)d_in[2];
  const float* Wq    = (const float*)d_in[3];
  const float* bq    = (const float*)d_in[4];
  const float* Wk    = (const float*)d_in[5];
  const float* bk    = (const float*)d_in[6];
  const float* Wv    = (const float*)d_in[7];
  const float* bv    = (const float*)d_in[8];
  const float* rel   = (const float*)d_in[9];

  const size_t per = (size_t)B_SZ * NUM_HEADS * L_PAD * D_K;
  u16* Qb = (u16*)d_ws;
  u16* Kb = Qb + per;
  u16* Vt = Kb + per;

  (void)hipMemsetAsync(d_ws, 0, 3 * per * sizeof(u16), stream);

  dim3 pgrid((B_SZ * L_SEQ) / 16, (D_MODEL / 16) / 4, 3);
  proj_kernel<<<pgrid, 128, 0, stream>>>(query, key, value, Wq, Wk, Wv,
                                         bq, bk, bv, Qb, Kb, Vt);

  dim3 agrid((L_PAD / 16) / 8, B_SZ * NUM_HEADS);
  attn_kernel<<<agrid, 256, 0, stream>>>(Qb, Kb, Vt, rel, (float*)d_out);
}
